// VQDecoderO2M_59691455480149
// MI455X (gfx1250) — compile-verified, matrix operand order checked by reading, not run on hardware
//
#include <hip/hip_runtime.h>
#include <hip/hip_bf16.h>

// CDNA5 / gfx1250 persistent LSTM-rollout kernel, v3 (branch-free activations).
// 64 workgroups x 512 threads (16 wave32). Each WG owns 64 batch rows for the
// whole sequence. Weights live in LDS as bf16; gates + projection GEMMs use
// v_wmma_f32_16x16x32_bf16. Wave w owns hidden block (w&7)*16 and row-half
// (w>>3): per-wave accumulators are 4 gates x 2 M-tiles = 64 VGPRs (register-
// resident, verified no scratch spill), and cell state c stays in registers in
// exact WMMA C-fragment layout. Activations use v_exp_f32/v_rcp_f32 with no
// EXEC divergence.

typedef __attribute__((ext_vector_type(16))) __bf16 v16bf;
typedef __attribute__((ext_vector_type(8)))  float  v8f;
typedef __attribute__((ext_vector_type(4)))  unsigned int u32x4;

#define HIDDEN 128
#define D_IO   80          // STATE(64) + ACTION(16)
#define KZ     224         // 80 + 128 + 16 zero pad  (7 * 32)
#define ROWS   64          // batch rows per workgroup
#define NG     512         // 4*HIDDEN gate columns
#define NTHREADS 512

union FragU { u32x4 u[2]; v16bf v; };

static __device__ __forceinline__ unsigned short f32_to_bf16(float f) {
    unsigned int u = __float_as_uint(f);
    u += 0x7FFFu + ((u >> 16) & 1u);      // round to nearest even
    return (unsigned short)(u >> 16);
}

// Load one bf16 WMMA fragment: two contiguous 16B chunks, 32 bytes apart
// (the lane's K-half selection is already folded into elem_ptr).
static __device__ __forceinline__ v16bf ld_frag(const unsigned short* elem_ptr) {
    const u32x4* p = (const u32x4*)elem_ptr;
    FragU f;
    f.u[0] = p[0];
    f.u[1] = p[2];   // +32 bytes
    return f.v;
}

// Branch-free activations: one v_exp_f32 + one v_rcp_f32 each, no saveexec.
static __device__ __forceinline__ float sigm(float x) {
    return __builtin_amdgcn_rcpf(1.0f + __expf(-x));
}
static __device__ __forceinline__ float tanh_fast(float x) {
    float ax = __builtin_fabsf(x);
    float t  = __expf(-2.0f * ax);                       // underflows to 0 -> r = 1
    float r  = (1.0f - t) * __builtin_amdgcn_rcpf(1.0f + t);
    return __builtin_copysignf(r, x);
}

__global__ void lstm_rollout_wmma(const float* __restrict__ hn0,   // (4096,128)
                                  const float* __restrict__ W_ih,  // (512,80)
                                  const float* __restrict__ W_hh,  // (512,128)
                                  const float* __restrict__ b_ih,  // (512)
                                  const float* __restrict__ b_hh,  // (512)
                                  const float* __restrict__ W_fc,  // (80,128)
                                  const float* __restrict__ b_fc,  // (80)
                                  const int*   __restrict__ seqp,
                                  float* __restrict__ out)         // (4096,T,80)
{
    extern __shared__ char smem[];
    unsigned short* Wg   = (unsigned short*)smem;           // [512][224] bf16
    unsigned short* Wfc  = Wg   + NG * KZ;                  // [80][128]  bf16
    unsigned short* zbuf = Wfc  + D_IO * HIDDEN;            // [64][224]  bf16
    float*          bsum = (float*)(zbuf + ROWS * KZ);      // [512] f32
    float*          bfc  = bsum + NG;                       // [80]  f32

    const int tid   = threadIdx.x;
    const int lane  = tid & 31;
    const int wave  = tid >> 5;         // 0..15
    const int hb    = wave & 7;         // hidden block: cols hb*16..hb*16+15
    const int mhalf = wave >> 3;        // 0: rows 0..31, 1: rows 32..63
    const int ln    = lane & 15;        // M / N index within tile
    const int half  = lane >> 4;        // K-half selector
    const int row0  = blockIdx.x * ROWS;
    const int T     = seqp[0];

    // ---------------- one-time setup: weights -> LDS (bf16) ----------------
    for (int i = tid; i < NG * KZ; i += NTHREADS) {
        int r = i / KZ, c = i - r * KZ;
        float v = 0.0f;
        if (c < D_IO)               v = W_ih[r * D_IO + c];
        else if (c < D_IO + HIDDEN) v = W_hh[r * HIDDEN + (c - D_IO)];
        Wg[i] = f32_to_bf16(v);
    }
    for (int i = tid; i < D_IO * HIDDEN; i += NTHREADS) Wfc[i] = f32_to_bf16(W_fc[i]);
    for (int i = tid; i < NG;   i += NTHREADS) bsum[i] = b_ih[i] + b_hh[i];
    for (int i = tid; i < D_IO; i += NTHREADS) bfc[i]  = b_fc[i];
    // z_0 = [x=0 | h=quantized_hn | pad=0]
    for (int i = tid; i < ROWS * KZ; i += NTHREADS) {
        int r = i / KZ, c = i - r * KZ;
        float v = 0.0f;
        if (c >= D_IO && c < D_IO + HIDDEN) v = hn0[(size_t)(row0 + r) * HIDDEN + (c - D_IO)];
        zbuf[i] = f32_to_bf16(v);
    }

    // persistent cell state in exact C-fragment layout: cfrag[ml][v] is element
    // (row = (mhalf*2+ml)*16 + half*8 + v, hidden col = hb*16 + ln)
    v8f cfrag[2];
    for (int ml = 0; ml < 2; ++ml)
        for (int v = 0; v < 8; ++v) cfrag[ml][v] = 0.0f;

    __syncthreads();

    // ---------------------------- time loop ---------------------------------
    for (int step = 0; step < T; ++step) {
        // ===== phase 1: gates = z @ Wg^T + bias  (M=64, N=512, K=224) =====
        v8f acc[4][2];   // [gate][mtile-local]
        for (int g = 0; g < 4; ++g) {
            float bg = bsum[g * HIDDEN + hb * 16 + ln];
            for (int ml = 0; ml < 2; ++ml)
                for (int v = 0; v < 8; ++v) acc[g][ml][v] = bg;
        }
#pragma unroll 1
        for (int kb = 0; kb < KZ; kb += 32) {
            v16bf afr[2];
            for (int ml = 0; ml < 2; ++ml)
                afr[ml] = ld_frag(&zbuf[((mhalf * 2 + ml) * 16 + ln) * KZ + kb + half * 8]);
            for (int g = 0; g < 4; ++g) {
                v16bf bfr = ld_frag(&Wg[(g * HIDDEN + hb * 16 + ln) * KZ + kb + half * 8]);
                for (int ml = 0; ml < 2; ++ml)
                    acc[g][ml] = __builtin_amdgcn_wmma_f32_16x16x32_bf16(
                        false, afr[ml], false, bfr, (short)0, acc[g][ml], false, false);
            }
        }

        // ===== pointwise LSTM (wave-local; i,f,g,o share lane/slot) =====
        float hn[2][8];
        for (int ml = 0; ml < 2; ++ml)
            for (int v = 0; v < 8; ++v) {
                float iv = acc[0][ml][v];
                float fv = acc[1][ml][v];
                float gv = acc[2][ml][v];
                float ov = acc[3][ml][v];
                float cn = sigm(fv) * cfrag[ml][v] + sigm(iv) * tanh_fast(gv);
                cfrag[ml][v] = cn;
                hn[ml][v] = sigm(ov) * tanh_fast(cn);
            }

        __syncthreads();   // all zbuf reads of this step done
        for (int ml = 0; ml < 2; ++ml)
            for (int v = 0; v < 8; ++v)
                zbuf[((mhalf * 2 + ml) * 16 + half * 8 + v) * KZ + D_IO + hb * 16 + ln] =
                    f32_to_bf16(hn[ml][v]);
        __syncthreads();   // h_new visible

        // ===== phase 2: y = h_new @ W_fc^T + b_fc  (M=64, N=80, K=128) =====
        for (int t = wave; t < 20; t += 16) {
            int mt = t & 3;       // M-tile 0..3
            int nt = t >> 2;      // N-tile 0..4
            float bb = bfc[nt * 16 + ln];
            v8f a2;
            for (int v = 0; v < 8; ++v) a2[v] = bb;
#pragma unroll
            for (int kb = 0; kb < HIDDEN; kb += 32) {
                v16bf afr = ld_frag(&zbuf[(mt * 16 + ln) * KZ + D_IO + kb + half * 8]);
                v16bf bfr = ld_frag(&Wfc[(nt * 16 + ln) * HIDDEN + kb + half * 8]);
                a2 = __builtin_amdgcn_wmma_f32_16x16x32_bf16(
                    false, afr, false, bfr, (short)0, a2, false, false);
            }
            // scatter y: global f32 stream + feedback into zbuf x-columns (bf16)
            int n = nt * 16 + ln;
            for (int v = 0; v < 8; ++v) {
                int r = mt * 16 + half * 8 + v;
                out[(size_t)(row0 + r) * (size_t)T * D_IO + (size_t)step * D_IO + n] = a2[v];
                zbuf[r * KZ + n] = f32_to_bf16(a2[v]);
            }
        }
        __syncthreads();   // x feedback visible before next step's reads
    }
}

extern "C" void kernel_launch(void* const* d_in, const int* in_sizes, int n_in,
                              void* d_out, int out_size, void* d_ws, size_t ws_size,
                              hipStream_t stream) {
    const float* hn0  = (const float*)d_in[0];
    const float* W_ih = (const float*)d_in[1];
    const float* W_hh = (const float*)d_in[2];
    const float* b_ih = (const float*)d_in[3];
    const float* b_hh = (const float*)d_in[4];
    const float* W_fc = (const float*)d_in[5];
    const float* b_fc = (const float*)d_in[6];
    const int*   seqp = (const int*)d_in[7];
    float* out = (float*)d_out;

    int batch = in_sizes[0] / HIDDEN;          // 4096
    int nblocks = batch / ROWS;                // 64 persistent workgroups

    size_t shmem = (size_t)(NG * KZ + D_IO * HIDDEN + ROWS * KZ) * sizeof(unsigned short)
                 + (size_t)(NG + D_IO) * sizeof(float);   // ~274 KB of 320 KB LDS

    lstm_rollout_wmma<<<dim3(nblocks), dim3(NTHREADS), shmem, stream>>>(
        hn0, W_ih, W_hh, b_ih, b_hh, W_fc, b_fc, seqp, out);
}
